// GraphRNNEncoder_53566832115727
// MI455X (gfx1250) — compile-verified, matrix operand order checked by reading, not run on hardware
//
#include <hip/hip_runtime.h>
#include <math.h>

// ---------------- problem constants (match reference) ----------------
constexpr int cB = 8, cN = 2048, cK = 16, cD = 256, cH = 256;
constexpr int ROWS = cB * cN;      // 16384
constexpr int G4H  = 4 * cH;       // 1024

// ---------------- WMMA types ----------------
typedef __attribute__((ext_vector_type(16))) __bf16 v16bf;
typedef __attribute__((ext_vector_type(8)))  __bf16 v8bf;
typedef __attribute__((ext_vector_type(8)))  float  v8f;

union FragBF { v16bf v; v8bf h[2]; };

// Load one 16x32 bf16 fragment from an LDS tile stored row-major with
// `ldStride` halfword pitch.  CDNA5 16-bit A layout (ISA 7.12.2):
//   lanes 0-15  : row = lane,     halves 0-7 => K 0-7,   halves 8-15 => K 16-23
//   lanes 16-31 : row = lane-16,  halves 0-7 => K 8-15,  halves 8-15 => K 24-31
// B fragments use the same pattern with "row" = output column because the W
// tile is staged N-major (weights are pre-transposed once per call).
__device__ inline v16bf load_frag16x32(const __bf16* lds, int ldStride,
                                       int baseRow, int lane) {
  int lo = lane & 15, hi = lane >> 4;
  const __bf16* p = lds + (size_t)(baseRow + lo) * ldStride + hi * 8;
  FragBF f;
  f.h[0] = *(const v8bf*)(p);
  f.h[1] = *(const v8bf*)(p + 16);
  return f.v;
}

// Async 16-byte global -> LDS copy (per-lane), tracked with ASYNCcnt.
// ldsOff = low 32 bits of the generic shared-memory pointer (LDS aperture
// addresses carry the wave-relative LDS offset in [31:0]).
__device__ inline void async_copy_b128(unsigned ldsOff, const void* gptr) {
  unsigned long long ga = (unsigned long long)(uintptr_t)gptr;
  asm volatile("global_load_async_to_lds_b128 %0, %1, off"
               :: "v"(ldsOff), "v"(ga) : "memory");
}
__device__ inline void async_wait0() {
  asm volatile("s_wait_asynccnt 0" ::: "memory");
}

// ---------------- WMMA GEMM ----------------
// out[M,N] = act( sum_p A_p[M,K] @ Wt_p[N,K]^T  + base )
// Wt is the bf16 weight stored TRANSPOSED ([N][K] row-major).
// mode 0: out_bf16 = tanh(acc + baseVec[col])                (neighbor proj)
// mode 1: out_f32  = acc + baseVec?[col] + baseMat?[row,col] (gate pre-acts)
// Tile: 128(M) x 64(N) per workgroup, 8 waves in a 4x2 grid, K-step 32.
__global__ __launch_bounds__(256)
void gemm_wmma_kernel(const __bf16* __restrict__ A0, const __bf16* __restrict__ Wt0,
                      const __bf16* __restrict__ A1, const __bf16* __restrict__ Wt1,
                      const float* __restrict__ baseVec,
                      const float* __restrict__ baseMat,
                      float* __restrict__ outF, __bf16* __restrict__ outB,
                      int Kdim, int Ncols, int npairs, int mode) {
  constexpr int LDA = 40;   // 32 + 8 halfword pad (80B row pitch, 16B aligned segs)
  __shared__ __bf16 sA[128 * LDA];
  __shared__ __bf16 sB[64 * LDA];

  const int tid  = threadIdx.x;
  const int lane = tid & 31;
  const int w    = tid >> 5;
  const int wm   = (w & 3) * 32;     // wave row offset inside tile
  const int wn   = (w >> 2) * 32;    // wave col offset inside tile
  const int bm   = blockIdx.x * 128;
  const int bn   = blockIdx.y * 64;

  v8f acc[2][2];
  const v8f zero = {0.f, 0.f, 0.f, 0.f, 0.f, 0.f, 0.f, 0.f};
  for (int i = 0; i < 2; ++i)
    for (int j = 0; j < 2; ++j) acc[i][j] = zero;

  for (int p = 0; p < npairs; ++p) {
    const __bf16* A  = p ? A1  : A0;
    const __bf16* Wt = p ? Wt1 : Wt0;
    for (int k0 = 0; k0 < Kdim; k0 += 32) {
      __syncthreads();   // protect previous iteration's LDS reads
      {  // stage A tile: 128 rows x 32 K, 2 threads/row, 2x16B async DMA each
        int row = tid >> 1, seg = tid & 1;
        const __bf16* g = A + (size_t)(bm + row) * Kdim + k0 + seg * 16;
        unsigned l = (unsigned)(uintptr_t)&sA[row * LDA + seg * 16];
        async_copy_b128(l,      g);
        async_copy_b128(l + 16, g + 8);
      }
      {  // stage Wt tile: 64 cols(n) x 32 K, N-major, 4 threads/col, 16B each
        int n = tid >> 2, seg = tid & 3;
        const __bf16* g = Wt + (size_t)(bn + n) * Kdim + k0 + seg * 8;
        unsigned l = (unsigned)(uintptr_t)&sB[n * LDA + seg * 8];
        async_copy_b128(l, g);
      }
      async_wait0();     // own wave's async fills done
      __syncthreads();   // everyone's fills visible
#pragma unroll
      for (int mi = 0; mi < 2; ++mi) {
        v16bf a = load_frag16x32(sA, LDA, wm + mi * 16, lane);
#pragma unroll
        for (int ni = 0; ni < 2; ++ni) {
          v16bf b = load_frag16x32(sB, LDA, wn + ni * 16, lane);
          acc[mi][ni] = __builtin_amdgcn_wmma_f32_16x16x32_bf16(
              false, a, false, b, (short)0, acc[mi][ni], false, false);
        }
      }
    }
  }

  // Epilogue.  C/D layout: VGPR i -> row (i + 8*hi), lane lo -> col.
  const int lo = lane & 15, hi = lane >> 4;
#pragma unroll
  for (int mi = 0; mi < 2; ++mi) {
#pragma unroll
    for (int ni = 0; ni < 2; ++ni) {
      const int col = bn + wn + ni * 16 + lo;
#pragma unroll
      for (int i = 0; i < 8; ++i) {
        const int row = bm + wm + mi * 16 + hi * 8 + i;
        float v = acc[mi][ni][i];
        if (mode == 0) {
          v = tanhf(v + baseVec[col]);
          outB[(size_t)row * Ncols + col] = (__bf16)v;
        } else {
          if (baseVec) v += baseVec[col];
          if (baseMat) v += baseMat[(size_t)row * Ncols + col];
          outF[(size_t)row * Ncols + col] = v;
        }
      }
    }
  }
}

// ---------------- node embedding gather ----------------
__global__ void node_emb_kernel(const int* __restrict__ labels,
                                const float* __restrict__ nmask,
                                const float* __restrict__ word_emb,
                                float* __restrict__ outF,
                                __bf16* __restrict__ outB) {
  const int row = blockIdx.x, d = threadIdx.x;
  const float v = word_emb[(size_t)labels[row] * cD + d] * nmask[row];
  outF[(size_t)row * cD + d] = v;
  outB[(size_t)row * cD + d] = (__bf16)v;
}

// ---------------- f32 -> bf16, transposed ([K][N] -> [N][K]) ----------------
__global__ void f2bfT_kernel(const float* __restrict__ src,
                             __bf16* __restrict__ dst, int K, int N) {
  const int i = blockIdx.x * blockDim.x + threadIdx.x;
  if (i < K * N) {
    const int k = i / N, n = i % N;
    dst[(size_t)n * K + k] = (__bf16)src[i];
  }
}

// ---------------- neighbor embedding gather+sum ----------------
// x[row, 0:D]   = sum_k edge_emb[lbl_k] * m_k      (f32 table, L2 resident)
// x[row, D:2D]  = sum_k node_emb[b, idx_k] * m_k   (bf16 gather, halves HBM bytes)
__global__ void nbr_embed_kernel(const float* __restrict__ edge_emb,
                                 const __bf16* __restrict__ node_bf,
                                 const int* __restrict__ eLbl,
                                 const int* __restrict__ nIdx,
                                 const float* __restrict__ nmK,
                                 __bf16* __restrict__ x) {
  const int row = blockIdx.x, d = threadIdx.x;
  const int b = row / cN;
  float ae = 0.f, an = 0.f;
#pragma unroll
  for (int k = 0; k < cK; ++k) {
    const int   lbl = eLbl[row * cK + k];
    const int   idx = nIdx[row * cK + k];
    const float m   = nmK[row * cK + k];
    ae += edge_emb[(size_t)lbl * cD + d] * m;
    an += (float)node_bf[((size_t)b * cN + idx) * cD + d] * m;
  }
  x[(size_t)row * (2 * cD) + d]      = (__bf16)ae;
  x[(size_t)row * (2 * cD) + cD + d] = (__bf16)an;
}

// ---------------- hidden-state gather+sum ----------------
__global__ void hid_gather_kernel(const __bf16* __restrict__ h_bf,
                                  const int* __restrict__ nIdx,
                                  const float* __restrict__ nmK,
                                  const float* __restrict__ nodes_mask,
                                  __bf16* __restrict__ xh) {
  const int row = blockIdx.x, d = threadIdx.x;
  const int b = row / cN;
  float a = 0.f;
#pragma unroll
  for (int k = 0; k < cK; ++k) {
    const int   idx = nIdx[row * cK + k];
    const float m   = nmK[row * cK + k];
    a += (float)h_bf[((size_t)b * cN + idx) * cH + d] * m;
  }
  xh[(size_t)row * cH + d] = (__bf16)(a * nodes_mask[row]);
}

// ---------------- fused LSTM gate / state update ----------------
__global__ void gate_kernel(const float* __restrict__ g,
                            const float* __restrict__ nodes_mask,
                            float* __restrict__ c,
                            float* __restrict__ out_h,
                            __bf16* __restrict__ h_bf,
                            float* __restrict__ final_h, int first) {
  const int row = blockIdx.x, j = threadIdx.x;
  const float* gr = g + (size_t)row * G4H;
  const float ig = 1.f / (1.f + __expf(-gr[j]));
  const float og = 1.f / (1.f + __expf(-gr[j + cH]));
  const float fg = 1.f / (1.f + __expf(-gr[j + 2 * cH]));
  const float ci = tanhf(gr[j + 3 * cH]);
  const size_t o = (size_t)row * cH + j;
  const float cp = first ? 0.f : c[o];
  const float nm = nodes_mask[row];
  const float cn = (fg * cp + ig * ci) * nm;
  const float hn = og * tanhf(cn);
  c[o]     = cn;
  out_h[o] = hn;
  h_bf[o]  = (__bf16)hn;
  if (final_h) final_h[o] = hn;
}

// ---------------- launch ----------------
extern "C" void kernel_launch(void* const* d_in, const int* in_sizes, int n_in,
                              void* d_out, int out_size, void* d_ws, size_t ws_size,
                              hipStream_t stream) {
  const int*   node_labels = (const int*)d_in[0];
  const float* nodes_mask  = (const float*)d_in[1];
  const int*   in_eLbl  = (const int*)d_in[2];
  const int*   in_nIdx  = (const int*)d_in[3];
  const float* in_mK    = (const float*)d_in[4];
  const int*   out_eLbl = (const int*)d_in[5];
  const int*   out_nIdx = (const int*)d_in[6];
  const float* out_mK   = (const float*)d_in[7];
  const float* word_emb = (const float*)d_in[8];
  const float* edge_emb = (const float*)d_in[9];
  const float* Wn   = (const float*)d_in[10];
  const float* bn   = (const float*)d_in[11];
  const float* Win  = (const float*)d_in[12];
  const float* Wout = (const float*)d_in[13];
  const float* Uin  = (const float*)d_in[14];
  const float* Uout = (const float*)d_in[15];
  const float* bg   = (const float*)d_in[16];

  // d_out layout: outs[3,ROWS,H] | node_emb[ROWS,D] | h[ROWS,H] | c[ROWS,H]
  float* outs         = (float*)d_out;
  float* node_emb_out = outs + (size_t)3 * ROWS * cH;
  float* h_out        = node_emb_out + (size_t)ROWS * cD;
  float* c_out        = h_out + (size_t)ROWS * cH;   // c persists across layers here

  // workspace carve-out
  char*  ws  = (char*)d_ws;
  size_t off = 0;
  auto alloc = [&](size_t bytes) -> void* {
    void* p = ws + off;
    off += bytes;
    off = (off + 255) & ~(size_t)255;
    return p;
  };
  __bf16* xin     = (__bf16*)alloc((size_t)ROWS * 2 * cD * 2);
  __bf16* xout    = (__bf16*)alloc((size_t)ROWS * 2 * cD * 2);
  __bf16* node_bf = (__bf16*)alloc((size_t)ROWS * cD * 2);
  __bf16* ine_bf  = (__bf16*)alloc((size_t)ROWS * cH * 2);
  __bf16* oute_bf = (__bf16*)alloc((size_t)ROWS * cH * 2);
  __bf16* ih_bf   = (__bf16*)alloc((size_t)ROWS * cH * 2);
  __bf16* oh_bf   = (__bf16*)alloc((size_t)ROWS * cH * 2);
  __bf16* h_bf    = (__bf16*)alloc((size_t)ROWS * cH * 2);
  __bf16* WnT_bf   = (__bf16*)alloc((size_t)2 * cD * cH * 2);   // [H][2D]
  __bf16* WinT_bf  = (__bf16*)alloc((size_t)cH * G4H * 2);      // [4H][H]
  __bf16* WoutT_bf = (__bf16*)alloc((size_t)cH * G4H * 2);
  __bf16* UinT_bf  = (__bf16*)alloc((size_t)cH * G4H * 2);
  __bf16* UoutT_bf = (__bf16*)alloc((size_t)cH * G4H * 2);
  float*  pre     = (float*)alloc((size_t)ROWS * G4H * 4);
  float*  g       = (float*)alloc((size_t)ROWS * G4H * 4);

  // 1) node embedding gather (f32 output slot + bf16 working copy)
  node_emb_kernel<<<ROWS, cD, 0, stream>>>(node_labels, nodes_mask, word_emb,
                                           node_emb_out, node_bf);

  // 2) weights -> bf16, transposed to [N][K] so GEMM B-staging is contiguous
  {
    int n1 = 2 * cD * cH;   // Wn: K=512, N=256
    int n2 = cH * G4H;      // Win/Wout/Uin/Uout: K=256, N=1024
    f2bfT_kernel<<<(n1 + 255) / 256, 256, 0, stream>>>(Wn,   WnT_bf,   2 * cD, cH);
    f2bfT_kernel<<<(n2 + 255) / 256, 256, 0, stream>>>(Win,  WinT_bf,  cH, G4H);
    f2bfT_kernel<<<(n2 + 255) / 256, 256, 0, stream>>>(Wout, WoutT_bf, cH, G4H);
    f2bfT_kernel<<<(n2 + 255) / 256, 256, 0, stream>>>(Uin,  UinT_bf,  cH, G4H);
    f2bfT_kernel<<<(n2 + 255) / 256, 256, 0, stream>>>(Uout, UoutT_bf, cH, G4H);
  }

  // 3) neighbor-embedding gathers -> bf16 A-matrices [ROWS, 2D]
  nbr_embed_kernel<<<ROWS, cD, 0, stream>>>(edge_emb, node_bf, in_eLbl, in_nIdx,
                                            in_mK, xin);
  nbr_embed_kernel<<<ROWS, cD, 0, stream>>>(edge_emb, node_bf, out_eLbl, out_nIdx,
                                            out_mK, xout);

  // 4) in_e / out_e = tanh(x @ Wn + bn)    M=16384 K=512 N=256
  {
    dim3 grid(ROWS / 128, cH / 64);
    gemm_wmma_kernel<<<grid, 256, 0, stream>>>(xin, WnT_bf, nullptr, nullptr,
                                               bn, nullptr, nullptr, ine_bf,
                                               2 * cD, cH, 1, 0);
    gemm_wmma_kernel<<<grid, 256, 0, stream>>>(xout, WnT_bf, nullptr, nullptr,
                                               bn, nullptr, nullptr, oute_bf,
                                               2 * cD, cH, 1, 0);
  }

  // 5) pre_static = in_e @ Win + out_e @ Wout + bg    M=16384 K=256 N=1024
  {
    dim3 grid(ROWS / 128, G4H / 64);
    gemm_wmma_kernel<<<grid, 256, 0, stream>>>(ine_bf, WinT_bf, oute_bf, WoutT_bf,
                                               bg, nullptr, pre, nullptr,
                                               cH, G4H, 2, 1);
  }

  // 6) recurrence, L = 3.  Layer 0: h == 0 so g == pre_static (skip gather+GEMM).
  gate_kernel<<<ROWS, cH, 0, stream>>>(pre, nodes_mask, c_out,
                                       outs + (size_t)0 * ROWS * cH, h_bf,
                                       nullptr, 1);
  for (int l = 1; l < 3; ++l) {
    hid_gather_kernel<<<ROWS, cH, 0, stream>>>(h_bf, in_nIdx, in_mK, nodes_mask, ih_bf);
    hid_gather_kernel<<<ROWS, cH, 0, stream>>>(h_bf, out_nIdx, out_mK, nodes_mask, oh_bf);
    dim3 grid(ROWS / 128, G4H / 64);
    gemm_wmma_kernel<<<grid, 256, 0, stream>>>(ih_bf, UinT_bf, oh_bf, UoutT_bf,
                                               nullptr, pre, g, nullptr,
                                               cH, G4H, 2, 1);
    gate_kernel<<<ROWS, cH, 0, stream>>>(g, nodes_mask, c_out,
                                         outs + (size_t)l * ROWS * cH, h_bf,
                                         (l == 2) ? h_out : nullptr, 0);
  }
}
